// SocialA2C_63410897158334
// MI455X (gfx1250) — matrix operand order repaired and checked by reading, not compiled, not run on hardware
//
#include <hip/hip_runtime.h>
#include <hip/hip_bf16.h>

// ---------- WMMA helper types (CDNA5 / gfx1250, wave32) ----------
typedef _Float16 v16h __attribute__((ext_vector_type(16)));
typedef _Float16 h8v  __attribute__((ext_vector_type(8)));
typedef _Float16 h4v  __attribute__((ext_vector_type(4)));
typedef float    v8f  __attribute__((ext_vector_type(8)));

// A-matrix 16x32 f16 layout (ISA 7.12.2): lane&15 = row M; lanes<16 hold
// K = {0..7, 16..23}, lanes>=16 hold K = {8..15, 24..31}.
static __device__ __forceinline__ v16h load_a_f32(const float* X, int ld,
                                                  int row0, int k0, int lane) {
  const int row = lane & 15;
  const int kb  = (lane & 16) ? 8 : 0;
  const float* p = X + (size_t)(row0 + row) * (size_t)ld + (size_t)(k0 + kb);
  float4 l0 = *(const float4*)(p + 0);
  float4 l1 = *(const float4*)(p + 4);
  float4 h0 = *(const float4*)(p + 16);
  float4 h1 = *(const float4*)(p + 20);
  v16h a;
  a[0] = (_Float16)l0.x;  a[1] = (_Float16)l0.y;
  a[2] = (_Float16)l0.z;  a[3] = (_Float16)l0.w;
  a[4] = (_Float16)l1.x;  a[5] = (_Float16)l1.y;
  a[6] = (_Float16)l1.z;  a[7] = (_Float16)l1.w;
  a[8] = (_Float16)h0.x;  a[9] = (_Float16)h0.y;
  a[10] = (_Float16)h0.z; a[11] = (_Float16)h0.w;
  a[12] = (_Float16)h1.x; a[13] = (_Float16)h1.y;
  a[14] = (_Float16)h1.z; a[15] = (_Float16)h1.w;
  return a;
}

// f16 source: two contiguous 16-byte segments per lane.
static __device__ __forceinline__ v16h load_a_f16(const _Float16* X, int ld,
                                                  int row0, int k0, int lane) {
  const int row = lane & 15;
  const int kb  = (lane & 16) ? 8 : 0;
  const _Float16* p = X + (size_t)(row0 + row) * (size_t)ld + (size_t)(k0 + kb);
  h8v lo = *(const h8v*)(p + 0);
  h8v hi = *(const h8v*)(p + 16);
  v16h a;
#pragma unroll
  for (int e = 0; e < 8; ++e) {
    a[e]     = lo[e];
    a[8 + e] = hi[e];
  }
  return a;
}

static __device__ __forceinline__ v16h load_a_lds(const _Float16* S, int ld,
                                                  int k0, int lane) {
  const int row = lane & 15;
  const int kb  = (lane & 16) ? 8 : 0;
  const _Float16* p = S + row * ld + k0 + kb;
  h8v lo = *(const h8v*)(p + 0);
  h8v hi = *(const h8v*)(p + 16);
  v16h a;
#pragma unroll
  for (int e = 0; e < 8; ++e) {
    a[e]     = lo[e];
    a[8 + e] = hi[e];
  }
  return a;
}

// B-matrix 32x16 f16 layout: lane&15 = col N; lanes<16 hold K={0..7,16..23},
// lanes>=16 hold K={8..15,24..31} (elements walk K, stride = ld).
static __device__ __forceinline__ v16h load_b_f32(const float* W, int ld,
                                                  int k0, int n0, int lane) {
  const int col = lane & 15;
  const int kb  = (lane & 16) ? 8 : 0;
  const float* p = W + (size_t)(k0 + kb) * (size_t)ld + (size_t)(n0 + col);
  v16h b;
#pragma unroll
  for (int e = 0; e < 16; ++e) b[e] = (_Float16)p[(size_t)(e < 8 ? e : e + 8) * (size_t)ld];
  return b;
}

static __device__ __forceinline__ v8f wmma_f16(v16h a, v16h b, v8f c) {
  return __builtin_amdgcn_wmma_f32_16x16x32_f16(false, a, false, b,
                                                (short)0, c, false, false);
}

// ---------- small utility kernels ----------
__global__ void fill_kernel(float* p, float v, long long n) {
  long long i = (long long)blockIdx.x * blockDim.x + threadIdx.x;
  long long stride = (long long)gridDim.x * blockDim.x;
  for (; i < n; i += stride) p[i] = v;
}

__global__ void degree_kernel(const int* ei, float* deg, int E_) {
  int i = blockIdx.x * blockDim.x + threadIdx.x;
  int stride = gridDim.x * blockDim.x;
  for (; i < E_; i += stride) {
    int d = ei[E_ + i];  // edge_index[1][i]
    atomicAdd(&deg[d], 1.0f);
  }
}

__global__ void rsqrt_kernel(const float* deg, float* dinv, int n) {
  int i = blockIdx.x * blockDim.x + threadIdx.x;
  if (i < n) {
    float d = deg[i];
    dinv[i] = (d > 0.0f) ? rsqrtf(d) : 0.0f;
  }
}

// h = relu(agg + bias), stored as f16 for downstream WMMA A-operands.
__global__ void bias_relu_f16_kernel(const float* x, const float* b,
                                     _Float16* y, long long total, int cols) {
  long long i = (long long)blockIdx.x * blockDim.x + threadIdx.x;
  long long stride = (long long)gridDim.x * blockDim.x;
  for (; i < total; i += stride) {
    float v = x[i] + b[(int)(i % cols)];
    y[i] = (_Float16)fmaxf(v, 0.0f);
  }
}

// agg[dst,:] += xw[src,:] * dinv[src]*dinv[dst]  (E edges + N self-loops)
// one thread per (edge, 4-half chunk); 8 chunks cover the 32 features.
// xw is f16 -> halves the dominant random-gather traffic.
__global__ void scatter_kernel(const int* ei, const _Float16* xw,
                               const float* dinv, float* agg, int E_, int N_) {
  long long total = (long long)(E_ + N_) * 8;
  long long i = (long long)blockIdx.x * blockDim.x + threadIdx.x;
  long long stride = (long long)gridDim.x * blockDim.x;
  for (; i < total; i += stride) {
    int e  = (int)(i >> 3);
    int ch = (int)(i & 7);
    int s, d;
    if (e < E_) {
      s = ei[e];
      d = ei[E_ + e];
    } else {
      s = d = e - E_;  // self-loop
    }
    float w = dinv[s] * dinv[d];
    h4v v = *(const h4v*)(xw + (size_t)s * 32 + ch * 4);
    float* o = agg + (size_t)d * 32 + ch * 4;
    atomicAdd(o + 0, (float)v[0] * w);
    atomicAdd(o + 1, (float)v[1] * w);
    atomicAdd(o + 2, (float)v[2] * w);
    atomicAdd(o + 3, (float)v[3] * w);
  }
}

// ---------- dense GEMMs: C[rows x ncols](f16) = A[rows x K] @ B[K x ncols] ----
// one wave (32 threads) per 16-row tile; fp16 WMMA, fp32 accumulate.
__global__ __launch_bounds__(32) void gemm_f32a_kernel(const float* A,
                                                       const float* B,
                                                       _Float16* C, int rows,
                                                       int K, int ncols) {
  const int lane = threadIdx.x;
  const int row0 = blockIdx.x * 16;
  if (row0 >= rows) return;
  const int n  = lane & 15;
  const int mb = (lane & 16) ? 8 : 0;
  const int arow = lane & 15;
  const int akb  = (lane & 16) ? 8 : 0;
  for (int n0 = 0; n0 < ncols; n0 += 16) {
    v8f c = {};
    for (int k0 = 0; k0 < K; k0 += 32) {
      if (k0 + 32 < K) {  // prefetch next A K-panel (global_prefetch_b8)
        __builtin_prefetch(A + (size_t)(row0 + arow) * (size_t)K +
                               (size_t)(k0 + 32 + akb), 0, 3);
      }
      v16h a = load_a_f32(A, K, row0, k0, lane);
      v16h b = load_b_f32(B, ncols, k0, n0, lane);
      c = wmma_f16(a, b, c);
    }
#pragma unroll
    for (int r = 0; r < 8; ++r)
      C[(size_t)(row0 + mb + r) * (size_t)ncols + (size_t)(n0 + n)] =
          (_Float16)c[r];
  }
}

__global__ __launch_bounds__(32) void gemm_f16a_kernel(const _Float16* A,
                                                       const float* B,
                                                       _Float16* C, int rows,
                                                       int K, int ncols) {
  const int lane = threadIdx.x;
  const int row0 = blockIdx.x * 16;
  if (row0 >= rows) return;
  const int n  = lane & 15;
  const int mb = (lane & 16) ? 8 : 0;
  for (int n0 = 0; n0 < ncols; n0 += 16) {
    v8f c = {};
    for (int k0 = 0; k0 < K; k0 += 32) {
      v16h a = load_a_f16(A, K, row0, k0, lane);
      v16h b = load_b_f32(B, ncols, k0, n0, lane);
      c = wmma_f16(a, b, c);
    }
#pragma unroll
    for (int r = 0; r < 8; ++r)
      C[(size_t)(row0 + mb + r) * (size_t)ncols + (size_t)(n0 + n)] =
          (_Float16)c[r];
  }
}

// ---------- fused A2C head ----------
// per wave: h tile [16x32](f16) -> hidden_p/hidden_v [16x512] (relu, f16, LDS)
// -> policy [16x64] via WMMA, value [16x1] via per-lane dot product.
#define HEAD_P 512
__global__ __launch_bounds__(32) void head_kernel(
    const _Float16* h, const float* pW1, const float* pb1, const float* pW2,
    const float* pb2, const float* vW1, const float* vb1, const float* vW2,
    const float* vb2, float* policy, float* value, int rows) {
  __shared__ _Float16 hp[16 * HEAD_P];
  __shared__ _Float16 hv[16 * HEAD_P];

  const int lane = threadIdx.x;
  const int row0 = blockIdx.x * 16;
  if (row0 >= rows) return;
  const int n  = lane & 15;
  const int mb = (lane & 16) ? 8 : 0;

  // A-tile of h (K=32) loaded once, reused for 64 WMMAs.
  v16h a0 = load_a_f16(h, 32, row0, 0, lane);

  // hidden_p = relu(h @ pW1 + pb1), staged to LDS as f16 row-major.
  for (int nt = 0; nt < HEAD_P / 16; ++nt) {
    v16h b = load_b_f32(pW1, HEAD_P, 0, nt * 16, lane);
    v8f c = {};
    c = wmma_f16(a0, b, c);
    float bias = pb1[nt * 16 + n];
#pragma unroll
    for (int r = 0; r < 8; ++r)
      hp[(mb + r) * HEAD_P + nt * 16 + n] = (_Float16)fmaxf(c[r] + bias, 0.0f);
  }
  // hidden_v = relu(h @ vW1 + vb1)
  for (int nt = 0; nt < HEAD_P / 16; ++nt) {
    v16h b = load_b_f32(vW1, HEAD_P, 0, nt * 16, lane);
    v8f c = {};
    c = wmma_f16(a0, b, c);
    float bias = vb1[nt * 16 + n];
#pragma unroll
    for (int r = 0; r < 8; ++r)
      hv[(mb + r) * HEAD_P + nt * 16 + n] = (_Float16)fmaxf(c[r] + bias, 0.0f);
  }
  __syncthreads();

  // policy = hidden_p @ pW2 + pb2   (K = 512, 4 N-tiles of 16 -> 64 cols)
  v8f acc0 = {}, acc1 = {}, acc2 = {}, acc3 = {};
  for (int ks = 0; ks < HEAD_P / 32; ++ks) {
    v16h a2 = load_a_lds(hp, HEAD_P, ks * 32, lane);
    acc0 = wmma_f16(a2, load_b_f32(pW2, 64, ks * 32, 0, lane), acc0);
    acc1 = wmma_f16(a2, load_b_f32(pW2, 64, ks * 32, 16, lane), acc1);
    acc2 = wmma_f16(a2, load_b_f32(pW2, 64, ks * 32, 32, lane), acc2);
    acc3 = wmma_f16(a2, load_b_f32(pW2, 64, ks * 32, 48, lane), acc3);
  }
#pragma unroll
  for (int r = 0; r < 8; ++r) {
    size_t base = (size_t)(row0 + mb + r) * 64;
    policy[base + 0 + n]  = acc0[r] + pb2[0 + n];
    policy[base + 16 + n] = acc1[r] + pb2[16 + n];
    policy[base + 32 + n] = acc2[r] + pb2[32 + n];
    policy[base + 48 + n] = acc3[r] + pb2[48 + n];
  }

  // value = hidden_v @ vW2 + vb2  (single output column: per-lane dot)
  if (lane < 16) {
    float s = vb2[0];
    for (int k = 0; k < HEAD_P; ++k)
      s += (float)hv[lane * HEAD_P + k] * vW2[k];
    value[row0 + lane] = s;
  }
}

// ---------- host-side orchestration ----------
static inline unsigned cdiv_u(long long a, long long b) {
  return (unsigned)((a + b - 1) / b);
}

extern "C" void kernel_launch(void* const* d_in, const int* in_sizes, int n_in,
                              void* d_out, int out_size, void* d_ws,
                              size_t ws_size, hipStream_t stream) {
  const float* features = (const float*)d_in[0];
  const int*   edges    = (const int*)d_in[1];   // int32 (JAX x64 disabled)
  const float* W1  = (const float*)d_in[2];
  const float* b1  = (const float*)d_in[3];
  const float* W2  = (const float*)d_in[4];
  const float* b2  = (const float*)d_in[5];
  const float* pW1 = (const float*)d_in[6];
  const float* pb1 = (const float*)d_in[7];
  const float* pW2 = (const float*)d_in[8];
  const float* pb2 = (const float*)d_in[9];
  const float* vW1 = (const float*)d_in[10];
  const float* vb1 = (const float*)d_in[11];
  const float* vW2 = (const float*)d_in[12];
  const float* vb2 = (const float*)d_in[13];

  const int H_ = in_sizes[3];           // 32
  const int F_ = in_sizes[2] / H_;      // 256
  const int N_ = in_sizes[0] / F_;      // 100000
  const int E_ = in_sizes[1] / 2;       // 1600000
  const int A_ = in_sizes[9];           // 64

  // workspace carve-up: f32 region first, then f16 region
  float* ws   = (float*)d_ws;
  float* deg  = ws;                               // N
  float* dinv = deg + N_;                         // N
  float* agg  = dinv + N_;                        // N*H
  _Float16* xw16 = (_Float16*)(agg + (size_t)N_ * H_);  // N*H halves
  _Float16* h16  = xw16 + (size_t)N_ * H_;              // N*H halves

  float* policy = (float*)d_out;
  float* value  = policy + (size_t)N_ * A_;

  const unsigned TB = 256;
  const unsigned gemm_blocks = cdiv_u(N_, 16);
  const unsigned scatter_blocks = cdiv_u((long long)(E_ + N_) * 8, TB);
  const unsigned elem_blocks = cdiv_u((long long)N_ * H_, TB);

  // degrees (self-loop => init 1.0) and symmetric norm
  fill_kernel<<<cdiv_u(N_, TB), TB, 0, stream>>>(deg, 1.0f, N_);
  degree_kernel<<<cdiv_u(E_, TB), TB, 0, stream>>>(edges, deg, E_);
  rsqrt_kernel<<<cdiv_u(N_, TB), TB, 0, stream>>>(deg, dinv, N_);

  // ---- conv1 ----
  gemm_f32a_kernel<<<gemm_blocks, 32, 0, stream>>>(features, W1, xw16, N_, F_, H_);
  fill_kernel<<<elem_blocks, TB, 0, stream>>>(agg, 0.0f, (long long)N_ * H_);
  scatter_kernel<<<scatter_blocks, TB, 0, stream>>>(edges, xw16, dinv, agg, E_, N_);
  bias_relu_f16_kernel<<<elem_blocks, TB, 0, stream>>>(agg, b1, h16,
                                                       (long long)N_ * H_, H_);
  // ---- conv2 ----
  gemm_f16a_kernel<<<gemm_blocks, 32, 0, stream>>>(h16, W2, xw16, N_, H_, H_);
  fill_kernel<<<elem_blocks, TB, 0, stream>>>(agg, 0.0f, (long long)N_ * H_);
  scatter_kernel<<<scatter_blocks, TB, 0, stream>>>(edges, xw16, dinv, agg, E_, N_);
  bias_relu_f16_kernel<<<elem_blocks, TB, 0, stream>>>(agg, b2, h16,
                                                       (long long)N_ * H_, H_);
  // ---- fused policy/value head ----
  head_kernel<<<gemm_blocks, 32, 0, stream>>>(h16, pW1, pb1, pW2, pb2, vW1,
                                              vb1, vW2, vb2, policy, value, N_);
}
